// QLSTM_65481071408196
// MI455X (gfx1250) — compile-verified, matrix operand order checked by reading, not run on hardware
//
#include <hip/hip_runtime.h>

// ---------------------------------------------------------------------------
// Persistent WMMA LSTM for MI455X (gfx1250, wave32).
//   T=1024, B=64, D=512, H=512.
//   Prep phase (throughput-bound, full chip):
//     - transpose/convert gate weights to bf16 (col-major slices)
//     - convert the whole input tensor fp32 -> bf16 (64 MB, L2-resident)
//   Scan phase (latency-bound): 32 persistent WGs x 256 threads (8 waves).
//     WG 'wg' owns hidden cols [wg*16, wg*16+16): 64 gate cols + 16 Wr cols.
//     Weights in LDS (bf16, col-major). Cell state in LDS. Hidden state
//     exchanged via double-buffered bf16 global buffer (128 KB, L2-resident).
//     One grid-wide generation barrier per timestep.
//   Wave roles: mt = w&3 (one M-tile per SIMD pair), nt0 = (w>>2)*2.
//   Waves 0-3 additionally run the Wr GEMM -> exactly one Wr-wave per SIMD.
//   All wave-role predicates are readfirstlane-scalarized so EXEC stays
//   all-ones across every WMMA (ISA requirement) and branches are s_cbranch.
// Workspace (~69 MB required):
//   WxT bf16 [4H][D]  (2 MB) | WhT bf16 [4H][H] (2 MB) | WrT bf16 [H][H] (.5 MB)
//   hbuf bf16 [2][B][H] (128 KB) | bar u32[2] | xbf bf16 [T][B][D] (64 MB)
// ---------------------------------------------------------------------------

#define T_STEPS 1024
#define BATCH   64
#define DIM     512
#define HID     512
#define NWG     32
#define CPW     16   // hidden columns per workgroup

typedef __attribute__((ext_vector_type(16))) __bf16       v16bf;
typedef __attribute__((ext_vector_type(8)))  float        v8f;
typedef __attribute__((ext_vector_type(4)))  unsigned int v4u;
typedef __attribute__((ext_vector_type(2)))  unsigned int v2u;
typedef __attribute__((ext_vector_type(4)))  float        v4f;

union BFrag {
  v16bf v;
  v4u   q[2];
  unsigned int u[8];
};

__device__ __forceinline__ unsigned short f2bf_u16(float x) {
  unsigned u = __float_as_uint(x);
  u += 0x7FFFu + ((u >> 16) & 1u);      // round-to-nearest-even
  return (unsigned short)(u >> 16);
}
__device__ __forceinline__ unsigned pack_bf2(float lo, float hi) {
  return (unsigned)f2bf_u16(lo) | ((unsigned)f2bf_u16(hi) << 16);
}

// sigmoid(2x) and tanh(y) via v_exp_f32 + v_rcp_f32 (cheap VALU tail)
__device__ __forceinline__ float fast_sig2(float x) {
  return __builtin_amdgcn_rcpf(1.0f + __expf(-2.0f * x));
}
__device__ __forceinline__ float fast_tanh(float y) {
  return 1.0f - 2.0f * __builtin_amdgcn_rcpf(__expf(2.0f * y) + 1.0f);
}

__device__ __forceinline__ v8f wmma_bf16(v16bf a, v16bf b, v8f c) {
  // 8 args: (neg_a, A, neg_b, B, c_mod, C, reuse_a, reuse_b)
  return __builtin_amdgcn_wmma_f32_16x16x32_bf16(false, a, false, b,
                                                 (short)0, c, false, false);
}

// B fragment (K=32 x N=16) from LDS, column-major base[col*512 + k].
// Lane map (mirrors 16-bit A map): lane L -> col = n0 + (L&15),
// first 8 K at k0 + ((L>>4)*8), second 8 K at +16.
__device__ __forceinline__ v16bf lds_fragB(const unsigned short* base,
                                           int n0, int k0, int lane) {
  int col = n0 + (lane & 15);
  int kh  = k0 + ((lane >> 4) << 3);
  const unsigned short* p = base + col * 512 + kh;
  BFrag f;
  f.q[0] = *(const v4u*)(p);        // K kh..kh+7
  f.q[1] = *(const v4u*)(p + 16);   // K kh+16..kh+23
  return f.v;
}

// A fragment (M=16 x K=32) from a row-major bf16 global matrix [.][512].
__device__ __forceinline__ v16bf glb_fragA_bf(const unsigned short* base,
                                              int m0, int k0, int lane) {
  int row = m0 + (lane & 15);
  int kh  = k0 + ((lane >> 4) << 3);
  const unsigned short* p = base + (size_t)row * 512 + kh;
  BFrag f;
  f.q[0] = *(const v4u*)(p);
  f.q[1] = *(const v4u*)(p + 16);
  return f.v;
}

// ---------------------------------------------------------------------------
// Prep 1: transpose/convert weights to bf16, zero hx buffers, reset barrier.
// Runs every launch (graph replays must be self-contained).
// ---------------------------------------------------------------------------
__global__ __launch_bounds__(256) void qlstm_prep(
    const float* __restrict__ Wf, const float* __restrict__ Wi,
    const float* __restrict__ Wg, const float* __restrict__ Wo,
    const float* __restrict__ Wr,
    unsigned short* __restrict__ WxT, unsigned short* __restrict__ WhT,
    unsigned short* __restrict__ WrT, unsigned short* __restrict__ hbuf,
    unsigned int* __restrict__ bar) {
  const int stride = gridDim.x * blockDim.x;
  const int gtid   = blockIdx.x * blockDim.x + threadIdx.x;

  for (int i = gtid; i < 4 * HID * DIM; i += stride) {
    int g   = i >> 18;                 // / (512*512)
    int rem = i & (HID * DIM - 1);
    int n   = rem >> 9;                // gate column
    int k   = rem & (DIM - 1);         // reduction index
    const float* W = (g == 0) ? Wf : (g == 1) ? Wi : (g == 2) ? Wg : Wo;
    WxT[i] = f2bf_u16(W[k * HID + n]);           // top D rows of W*
    WhT[i] = f2bf_u16(W[(DIM + k) * HID + n]);   // bottom H rows of W*
  }
  for (int i = gtid; i < HID * HID; i += stride) {
    int n = i >> 9, k = i & (HID - 1);
    WrT[i] = f2bf_u16(Wr[k * HID + n]);
  }
  for (int i = gtid; i < 2 * BATCH * HID; i += stride) hbuf[i] = 0;
  if (gtid == 0) { bar[0] = 0u; bar[1] = 0u; }
}

// ---------------------------------------------------------------------------
// Prep 2: stream-convert inputs fp32 -> bf16 (vectorized float4 -> 2x u32).
// Pure bandwidth pass; 64 MB result stays resident in the 192 MB L2.
// ---------------------------------------------------------------------------
__global__ __launch_bounds__(256) void qlstm_xconv(
    const float* __restrict__ x, unsigned short* __restrict__ xbf) {
  const size_t n4     = (size_t)T_STEPS * BATCH * DIM / 4;
  const size_t stride = (size_t)gridDim.x * blockDim.x;
  for (size_t i = (size_t)blockIdx.x * blockDim.x + threadIdx.x; i < n4;
       i += stride) {
    v4f a = ((const v4f*)x)[i];
    v2u o;
    o.x = pack_bf2(a.x, a.y);
    o.y = pack_bf2(a.z, a.w);
    ((v2u*)xbf)[i] = o;
  }
}

// ---------------------------------------------------------------------------
// Persistent scan kernel.
// ---------------------------------------------------------------------------
__global__ __launch_bounds__(256) void qlstm_scan(
    const unsigned short* __restrict__ xbf,       // [T,B,D] bf16
    const float* __restrict__ bf_, const float* __restrict__ bi_,
    const float* __restrict__ bg_, const float* __restrict__ bo_,
    const float* __restrict__ br_,
    const unsigned short* __restrict__ WxT,
    const unsigned short* __restrict__ WhT,
    const unsigned short* __restrict__ WrT,
    unsigned short* __restrict__ hbuf,            // [2][B][H] bf16
    unsigned int* __restrict__ bar,
    float* __restrict__ out) {
  extern __shared__ char smem[];
  unsigned short* WxL = (unsigned short*)smem;     // [64 cols][512 k] bf16
  unsigned short* WhL = WxL + 64 * 512;            // [64 cols][512 k]
  unsigned short* WrL = WhL + 64 * 512;            // [16 cols][512 k]
  float* preL = (float*)(WrL + 16 * 512);          // [64 m][64 gate-cols]
  float* rL   = preL + 64 * 64;                    // [64 m][16 cols]
  float* cL   = rL + 64 * 16;                      // [64 m][16 cols]

  const int wg   = blockIdx.x;
  const int tid  = threadIdx.x;
  const int w    = tid >> 5;
  const int lane = tid & 31;

  // ---- stage this WG's weight slices into LDS (once) ----
  for (int i = tid; i < 64 * 64; i += 256) {       // 64 cols x 64 uint4/row
    int cl = i >> 6, part = i & 63;
    int gate = cl >> 4, j = cl & 15;
    int rsrc = gate * HID + wg * CPW + j;
    ((v4u*)WxL)[cl * 64 + part] = ((const v4u*)WxT)[(size_t)rsrc * 64 + part];
    ((v4u*)WhL)[cl * 64 + part] = ((const v4u*)WhT)[(size_t)rsrc * 64 + part];
  }
  for (int i = tid; i < 16 * 64; i += 256) {
    int cl = i >> 6, part = i & 63;
    int rsrc = wg * CPW + cl;
    ((v4u*)WrL)[i] = ((const v4u*)WrT)[(size_t)rsrc * 64 + part];
  }
  for (int i = tid; i < BATCH * CPW; i += 256) cL[i] = 0.0f;
  __syncthreads();

  // Wave roles (readfirstlane-scalarized -> SGPR-uniform, s_cbranch only):
  //   mt  = w & 3  : M-tile; one per SIMD, so Wr-waves spread across SIMDs
  //   nt0 = (w>>2)*2 : N-tile pair {0,1}=f,i (waves 0-3) or {2,3}=g,o (4-7)
  //   doR = (w < 4): waves 0-3 also run the Wr GEMM (1 Wr-wave per SIMD)
  const int mt_s  = __builtin_amdgcn_readfirstlane(w & 3);
  const int nt0_s = __builtin_amdgcn_readfirstlane((w >> 2) * 2);
  const int doR_s = __builtin_amdgcn_readfirstlane((w < 4) ? 1 : 0);
  const int m0  = mt_s * 16;
  const int nt0 = nt0_s;
  const int colg_bias = wg * CPW + (lane & 15);
  const float bias0 = (nt0 == 0) ? bf_[colg_bias] : bg_[colg_bias];
  const float bias1 = (nt0 == 0) ? bi_[colg_bias] : bo_[colg_bias];

  const size_t OUT_HX = (size_t)T_STEPS * BATCH * HID;
  const size_t OUT_CX = OUT_HX + (size_t)BATCH * HID;

  for (int t = 0; t < T_STEPS; ++t) {
    const unsigned short* xrow  = xbf + (size_t)t * BATCH * DIM;
    const unsigned short* hprev = hbuf + (size_t)(t & 1) * BATCH * HID;
    unsigned short* hnext       = hbuf + (size_t)((t + 1) & 1) * BATCH * HID;

    // prefetch x_{t+1} (65536 B bf16 / 256 threads = 256 B each)
    if (t + 1 < T_STEPS) {
      const char* pf = (const char*)(xbf + (size_t)(t + 1) * BATCH * DIM);
      __builtin_prefetch(pf + tid * 256, 0, 0);
      __builtin_prefetch(pf + tid * 256 + 128, 0, 0);
    }

    v8f acc0, acc1, accR;
#pragma unroll
    for (int v = 0; v < 8; ++v) { acc0[v] = bias0; acc1[v] = bias1; accR[v] = 0.f; }

    // x_t @ Wx (K = 512 over D) -- straight b128 A loads, no conversion
#pragma unroll 4
    for (int k0 = 0; k0 < DIM; k0 += 32) {
      v16bf a  = glb_fragA_bf(xrow, m0, k0, lane);
      v16bf b0 = lds_fragB(WxL, nt0 * 16,      k0, lane);
      v16bf b1 = lds_fragB(WxL, nt0 * 16 + 16, k0, lane);
      acc0 = wmma_bf16(a, b0, acc0);
      acc1 = wmma_bf16(a, b1, acc1);
    }
    // h_{t-1} @ Wh (K = 512 over H); Wr-waves take a uniform scalar branch
    if (doR_s) {
#pragma unroll 4
      for (int k0 = 0; k0 < HID; k0 += 32) {
        v16bf a  = glb_fragA_bf(hprev, m0, k0, lane);
        v16bf b0 = lds_fragB(WhL, nt0 * 16,      k0, lane);
        v16bf b1 = lds_fragB(WhL, nt0 * 16 + 16, k0, lane);
        v16bf b2 = lds_fragB(WrL, 0,             k0, lane);
        acc0 = wmma_bf16(a, b0, acc0);
        acc1 = wmma_bf16(a, b1, acc1);
        accR = wmma_bf16(a, b2, accR);
      }
    } else {
#pragma unroll 4
      for (int k0 = 0; k0 < HID; k0 += 32) {
        v16bf a  = glb_fragA_bf(hprev, m0, k0, lane);
        v16bf b0 = lds_fragB(WhL, nt0 * 16,      k0, lane);
        v16bf b1 = lds_fragB(WhL, nt0 * 16 + 16, k0, lane);
        acc0 = wmma_bf16(a, b0, acc0);
        acc1 = wmma_bf16(a, b1, acc1);
      }
    }

    // dump accumulators to LDS (C/D layout: lane<16 -> M=v, else M=v+8; N=lane%16)
    {
      int ccol = lane & 15;
#pragma unroll
      for (int v = 0; v < 8; ++v) {
        int rrow = (lane < 16) ? v : (v + 8);
        preL[(m0 + rrow) * 64 + nt0 * 16 + ccol]       = acc0[v];
        preL[(m0 + rrow) * 64 + (nt0 + 1) * 16 + ccol] = acc1[v];
      }
      if (doR_s) {
#pragma unroll
        for (int v = 0; v < 8; ++v) {
          int rrow = (lane < 16) ? v : (v + 8);
          rL[(m0 + rrow) * 16 + ccol] = accR[v];
        }
      }
    }
    __syncthreads();

    // elementwise gate math + state update: 64m x 16cols = 1024 elems
#pragma unroll
    for (int e = 0; e < 4; ++e) {
      int idx = tid + 256 * e;
      int m = idx >> 4, j = idx & 15;
      int colg = wg * CPW + j;
      float fin = preL[m * 64 + j];
      float iin = preL[m * 64 + 16 + j];
      float gin = preL[m * 64 + 32 + j];
      float oin = preL[m * 64 + 48 + j];
      float fs = fast_sig2(fin);
      float is = fast_sig2(iin);
      float gs = fast_tanh(2.0f * gin);
      float os = fast_sig2(oin);
      float cold = cL[idx];
      float cnew = fs * cold + is * gs;
      float rv   = rL[m * 16 + j] + br_[colg];
      float hnew = os * fast_tanh(cnew) + 0.1f * rv;
      cL[idx] = cnew;
      out[(size_t)t * BATCH * HID + (size_t)m * HID + colg] = hnew;
      hnext[m * HID + colg] = f2bf_u16(hnew);
      if (t == T_STEPS - 1) {
        out[OUT_HX + (size_t)m * HID + colg] = hnew;
        out[OUT_CX + (size_t)m * HID + colg] = cnew;
      }
    }

    // ---- grid-wide generation barrier (monotonic counter, reset by prep) ----
    __threadfence();
    if (tid == 0) {
      unsigned gen = (unsigned)(t + 1);
      unsigned arrived = atomicAdd(&bar[0], 1u) + 1u;
      if (arrived == gen * NWG) {
        __threadfence();
        atomicExch(&bar[1], gen);
      } else {
        volatile unsigned* rel = &bar[1];
        while (*rel < gen) { __builtin_amdgcn_s_sleep(2); }
      }
    }
    __syncthreads();
    __threadfence();
  }
}

// ---------------------------------------------------------------------------
extern "C" void kernel_launch(void* const* d_in, const int* in_sizes, int n_in,
                              void* d_out, int out_size, void* d_ws, size_t ws_size,
                              hipStream_t stream) {
  const float* x  = (const float*)d_in[0];
  const float* Wf = (const float*)d_in[1];
  const float* bf = (const float*)d_in[2];
  const float* Wi = (const float*)d_in[3];
  const float* bi = (const float*)d_in[4];
  const float* Wg = (const float*)d_in[5];
  const float* bg = (const float*)d_in[6];
  const float* Wo = (const float*)d_in[7];
  const float* bo = (const float*)d_in[8];
  const float* Wr = (const float*)d_in[9];
  const float* br = (const float*)d_in[10];

  unsigned short* WxT  = (unsigned short*)d_ws;            // 4*512*512
  unsigned short* WhT  = WxT + 4 * HID * DIM;              // 4*512*512
  unsigned short* WrT  = WhT + 4 * HID * DIM;              // 512*512
  unsigned short* hbuf = WrT + HID * HID;                  // 2*64*512
  unsigned int*   bar  = (unsigned int*)(hbuf + 2 * BATCH * HID);
  unsigned short* xbf  = (unsigned short*)(bar + 16);      // 1024*64*512 (64 MB)

  qlstm_prep<<<1024, 256, 0, stream>>>(Wf, Wi, Wg, Wo, Wr,
                                       WxT, WhT, WrT, hbuf, bar);
  qlstm_xconv<<<4096, 256, 0, stream>>>(x, xbf);

  size_t lds_bytes = (size_t)(64 * 512 + 64 * 512 + 16 * 512) * sizeof(unsigned short)
                   + (size_t)(64 * 64 + 64 * 16 + 64 * 16) * sizeof(float); // 172032
  qlstm_scan<<<NWG, 256, lds_bytes, stream>>>(xbf, bf, bi, bg, bo, br,
                                              WxT, WhT, WrT, hbuf, bar,
                                              (float*)d_out);
}
